// FourierCrossAttention_45423574122668
// MI455X (gfx1250) — matrix operand order repaired and checked by reading, not compiled, not run on hardware
//
#include <hip/hip_runtime.h>

// ---------------------------------------------------------------------------
// FourierCrossAttention on gfx1250 (MI455X).
//   Stage 1: truncated forward DFT of q,k  -> f16 WMMA GEMM, K=4096
//   Stage 2: complex mode-space chain (tiny) -> f32 VALU in LDS
//   Stage 3: sparse inverse rFFT            -> f16 WMMA GEMM, K=128
// All normalization (1/4096 * 1/(512*512) = 2^-30) applied at final store.
// ---------------------------------------------------------------------------

typedef __attribute__((ext_vector_type(16))) _Float16 v16h;
typedef __attribute__((ext_vector_type(8)))  _Float16 v8h;
typedef __attribute__((ext_vector_type(8)))  float    v8f;

__device__ __forceinline__ v16h cat16(v8h lo, v8h hi) {
  return __builtin_shufflevector(lo, hi, 0,1,2,3,4,5,6,7,8,9,10,11,12,13,14,15);
}

#define NMODE   64
#define LQN     4096
#define EDIM    64
#define NBH     128          // B(16) * H(8)
#define KCHUNK  4            // forward-DFT K split (deterministic 2-pass reduce)
#define TWO_PI_OVER_N 1.5339807878856412e-3f  // 2*pi/4096
#define LDS_ROW 40           // halves per e-row in k_fwd staging (32 + pad, 80B)

// ---------------------------------------------------------------------------
// Twiddle tables (f16), laid out for contiguous WMMA fragment loads:
//   FcT[m][t] = cos(2*pi*m*t/N)        (B-frag of forward GEMM: K=t contiguous)
//   FsT[m][t] = -sin(2*pi*m*t/N)
//   GT [t][j] : j<64 -> c_m*cos, j>=64 -> -2*sin  (B-frag of inverse GEMM)
// ---------------------------------------------------------------------------
__global__ void k_tables(_Float16* __restrict__ FcT, _Float16* __restrict__ FsT,
                         _Float16* __restrict__ GT) {
  const int stride = gridDim.x * blockDim.x;
  const int t0 = blockIdx.x * blockDim.x + threadIdx.x;
  for (int i = t0; i < NMODE * LQN; i += stride) {
    int m = i >> 12, t = i & (LQN - 1);
    int r = (m * t) & (LQN - 1);            // exact angle mod 2*pi
    float th = (float)r * TWO_PI_OVER_N;
    FcT[i] = (_Float16)__cosf(th);
    FsT[i] = (_Float16)(-__sinf(th));
  }
  for (int i = t0; i < LQN * 128; i += stride) {
    int t = i >> 7, j = i & 127;
    int m = j & 63;
    int r = (m * t) & (LQN - 1);
    float th = (float)r * TWO_PI_OVER_N;
    float v = (j < 64) ? (((m == 0) ? 1.0f : 2.0f) * __cosf(th))
                       : (-2.0f * __sinf(th));
    GT[i] = (_Float16)v;
  }
}

// ---------------------------------------------------------------------------
// Forward truncated DFT:  Xpart[chunk][bh][comp][e][m] = sum_t x[t,e]*F[m][t]
// block = 256 thr (8 waves): wave -> (e_tile = wid&3, comp = wid>>2)
// grid  = NBH * KCHUNK
// ---------------------------------------------------------------------------
__global__ __launch_bounds__(256)
void k_fwd_dft(const float* __restrict__ x,
               const _Float16* __restrict__ FcT,
               const _Float16* __restrict__ FsT,
               float* __restrict__ Xpart) {
  __shared__ _Float16 xs[EDIM * LDS_ROW];   // [e][t] f16 tile, padded rows

  const int bid   = blockIdx.x;
  const int bh    = bid & (NBH - 1);
  const int chunk = bid >> 7;
  const int b = bh >> 3, h = bh & 7;
  const int tid = threadIdx.x, lane = tid & 31, wid = tid >> 5;
  const int e0 = (wid & 3) * 16;
  const int comp = wid >> 2;                 // 0 = cos(Re), 1 = -sin(Im)
  const int col = lane & 15;
  const int hs  = lane >> 4;                 // lane half-select
  const _Float16* __restrict__ Ftab = comp ? FsT : FcT;

  v8f acc[4] = {};                           // 4 mode tiles of 16

  const float* __restrict__ xb = x + (size_t)b * LQN * 512 + h * 64;
  const int kbeg = chunk * (LQN / KCHUNK);
  const int kend = kbeg + (LQN / KCHUNK);

  const int se = tid & 63;                   // staging: e
  const int stp = tid >> 6;                  // staging: t phase (0..3)

  for (int k0 = kbeg; k0 < kend; k0 += 32) {
    // ---- stage [32 t][64 e] f32 -> f16 LDS (coalesced over e) ----
#pragma unroll
    for (int i = 0; i < 8; ++i) {
      int t = stp + 4 * i;
      xs[se * LDS_ROW + t] = (_Float16)xb[(size_t)(k0 + t) * 512 + se];
    }
    __syncthreads();

    // ---- A fragment: A[e][t], lane M=e0+col, 16 contiguous K per lane ----
    const int eA = e0 + col;
    const int tb = hs * 16;
    v8h alo = *(const v8h*)&xs[eA * LDS_ROW + tb];        // K 0-7 / 16-23
    v8h ahi = *(const v8h*)&xs[eA * LDS_ROW + tb + 8];    // K 8-15 / 24-31
    v16h a = cat16(alo, ahi);

#pragma unroll
    for (int mt = 0; mt < 4; ++mt) {
      // B fragment: B[t][m] from FcT/FsT[m][t]; 16 contiguous K per lane
      size_t off = (size_t)(mt * 16 + col) * LQN + (size_t)k0 + hs * 16;
      v8h blo = *(const v8h*)&Ftab[off];
      v8h bhi = *(const v8h*)&Ftab[off + 8];
      v16h bb = cat16(blo, bhi);
      acc[mt] = __builtin_amdgcn_wmma_f32_16x16x32_f16(
          false, a, false, bb, (short)0, acc[mt], false, false);
    }
    __syncthreads();
  }

  // D layout: VGPR r -> M = r + 8*hs, N = col
  float* __restrict__ Xo =
      Xpart + (((size_t)chunk * NBH + bh) * 2 + comp) * (EDIM * NMODE);
#pragma unroll
  for (int mt = 0; mt < 4; ++mt)
#pragma unroll
    for (int r = 0; r < 8; ++r) {
      int e = e0 + hs * 8 + r;
      int m = mt * 16 + col;
      Xo[e * NMODE + m] = acc[mt][r];
    }
}

// ---------------------------------------------------------------------------
// Deterministic reduce of the KCHUNK partials for both Xq and Xk.
// ---------------------------------------------------------------------------
__global__ void k_reduce(const float* __restrict__ XqP,
                         const float* __restrict__ XkP,
                         float* __restrict__ Xq, float* __restrict__ Xk) {
  const int n = NBH * 2 * EDIM * NMODE;      // 1,048,576
  const int stride = gridDim.x * blockDim.x;
  for (int i = blockIdx.x * blockDim.x + threadIdx.x; i < n; i += stride) {
    float sq = 0.f, sk = 0.f;
#pragma unroll
    for (int c = 0; c < KCHUNK; ++c) {
      sq += XqP[(size_t)c * n + i];
      sk += XkP[(size_t)c * n + i];
    }
    Xq[i] = sq;
    Xk[i] = sk;
  }
}

// ---------------------------------------------------------------------------
// Mode-space chain per (b,h): xqk -> tanh -> xqkv -> *W, emit Y f16 [o][128]
// (<2% of total FLOPs: VALU + LDS, not worth WMMA fragment shuffles)
// ---------------------------------------------------------------------------
__global__ __launch_bounds__(256)
void k_modes(const float* __restrict__ Xq, const float* __restrict__ Xk,
             const float* __restrict__ w1, const float* __restrict__ w2,
             _Float16* __restrict__ Yh) {
  __shared__ float    sT[2 * 4096];          // tanh(xqk) [c][x][y]   (32 KB)
  __shared__ _Float16 sV[2 * 4096];          // xqkv      [c][e][x]   (16 KB)

  const int bh = blockIdx.x;
  const int h = bh & 7;
  const int tid = threadIdx.x;
  const float* __restrict__ q = Xq + (size_t)bh * 8192;   // [c][e][x]
  const float* __restrict__ kk = Xk + (size_t)bh * 8192;  // [c][e][y]

  // xqk[x][y] = sum_e Xq[e,x] * Xk[e,y]   (complex), then tanh(re),tanh(im)
  for (int i = tid; i < 4096; i += 256) {
    int xm = i >> 6, y = i & 63;
    float re = 0.f, im = 0.f;
    for (int e = 0; e < EDIM; ++e) {
      float qr = q[e * 64 + xm], qi = q[4096 + e * 64 + xm];
      float kr = kk[e * 64 + y], ki = kk[4096 + e * 64 + y];
      re += qr * kr - qi * ki;
      im += qr * ki + qi * kr;
    }
    sT[i] = tanhf(re);
    sT[4096 + i] = tanhf(im);
  }
  __syncthreads();

  // xqkv[e][x] = sum_y xqk[x,y] * Xk[e,y]  (complex)
  for (int i = tid; i < 4096; i += 256) {
    int e = i >> 6, xm = i & 63;
    float re = 0.f, im = 0.f;
    for (int y = 0; y < 64; ++y) {
      float tr = sT[xm * 64 + y], ti = sT[4096 + xm * 64 + y];
      float kr = kk[e * 64 + y], ki = kk[4096 + e * 64 + y];
      re += tr * kr - ti * ki;
      im += tr * ki + ti * kr;
    }
    sV[i] = (_Float16)re;
    sV[4096 + i] = (_Float16)im;
  }
  __syncthreads();

  // Y[o][x] = sum_e xqkv[e,x] * (w1 + i w2)[h,e,o,x]; pack [o][j], j=x / 64+x
  _Float16* __restrict__ yo = Yh + (size_t)bh * EDIM * 128;
  const float* __restrict__ w1h = w1 + (size_t)h * 64 * 64 * 64;
  const float* __restrict__ w2h = w2 + (size_t)h * 64 * 64 * 64;
  for (int i = tid; i < 4096; i += 256) {
    int o = i >> 6, xm = i & 63;
    float re = 0.f, im = 0.f;
    for (int e = 0; e < EDIM; ++e) {
      float vr = (float)sV[e * 64 + xm];
      float vi = (float)sV[4096 + e * 64 + xm];
      float wr = w1h[(e * 64 + o) * 64 + xm];
      float wi = w2h[(e * 64 + o) * 64 + xm];
      re += vr * wr - vi * wi;
      im += vr * wi + vi * wr;
    }
    yo[o * 128 + xm] = (_Float16)re;
    yo[o * 128 + 64 + xm] = (_Float16)im;
  }
}

// ---------------------------------------------------------------------------
// Sparse inverse rFFT: out[b,t,h,o] = S * sum_j Y[bh][o][j] * GT[t][j]
// grid = NBH * 128 (t-chunks of 32); 8 waves: (o_tile = wid&3, n_tile = wid>>2)
// ---------------------------------------------------------------------------
__global__ __launch_bounds__(256)
void k_inv_dft(const _Float16* __restrict__ Yh, const _Float16* __restrict__ GT,
               float* __restrict__ out) {
  const int bid = blockIdx.x;
  const int tc = bid & 127;
  const int bh = bid >> 7;
  const int b = bh >> 3, h = bh & 7;
  const int lane = threadIdx.x & 31, wid = threadIdx.x >> 5;
  const int o0 = (wid & 3) * 16;
  const int t0 = tc * 32 + (wid >> 2) * 16;
  const int col = lane & 15;
  const int hs = lane >> 4;

  v8f acc = {};
  const _Float16* __restrict__ ya = Yh + ((size_t)bh * EDIM + o0 + col) * 128;
  const _Float16* __restrict__ gb = GT + (size_t)(t0 + col) * 128;

#pragma unroll
  for (int k0 = 0; k0 < 128; k0 += 32) {
    v8h alo = *(const v8h*)&ya[k0 + hs * 16];       // A: 16 contiguous K
    v8h ahi = *(const v8h*)&ya[k0 + hs * 16 + 8];
    v16h a = cat16(alo, ahi);
    v8h blo = *(const v8h*)&gb[k0 + hs * 16];       // B: 16 contiguous K
    v8h bhi = *(const v8h*)&gb[k0 + hs * 16 + 8];
    v16h bb = cat16(blo, bhi);
    acc = __builtin_amdgcn_wmma_f32_16x16x32_f16(
        false, a, false, bb, (short)0, acc, false, false);
  }

  const float S = 1.0f / (4096.0f * 262144.0f);     // irfft 1/N * 1/(C_in*C_out)
  const int t = t0 + col;
  float* __restrict__ op =
      out + (((size_t)b * LQN + t) * 8 + h) * 64 + o0 + hs * 8;
#pragma unroll
  for (int r = 0; r < 8; ++r) op[r] = acc[r] * S;   // contiguous in o
}

// ---------------------------------------------------------------------------
extern "C" void kernel_launch(void* const* d_in, const int* in_sizes, int n_in,
                              void* d_out, int out_size, void* d_ws, size_t ws_size,
                              hipStream_t stream) {
  const float* q  = (const float*)d_in[0];
  const float* k  = (const float*)d_in[1];
  // d_in[2] (v) is computed-but-unused in the reference forward
  const float* w1 = (const float*)d_in[3];
  const float* w2 = (const float*)d_in[4];
  float* out = (float*)d_out;

  // workspace layout (bytes): 44 MB total
  char* ws = (char*)d_ws;
  float*    XqP = (float*)(ws);                          // 16 MB partials q
  float*    XkP = (float*)(ws + (16u << 20));            // 16 MB partials k
  float*    Xq  = (float*)(ws + (32u << 20));            //  4 MB
  float*    Xk  = (float*)(ws + (36u << 20));            //  4 MB
  _Float16* Yh  = (_Float16*)(ws + (40u << 20));         //  2 MB
  _Float16* FcT = (_Float16*)(ws + (42u << 20));         // 512 KB
  _Float16* FsT = (_Float16*)(ws + (42u << 20) + (512u << 10)); // 512 KB
  _Float16* GT  = (_Float16*)(ws + (43u << 20));         //  1 MB

  k_tables<<<256, 256, 0, stream>>>(FcT, FsT, GT);
  k_fwd_dft<<<NBH * KCHUNK, 256, 0, stream>>>(q, FcT, FsT, XqP);
  k_fwd_dft<<<NBH * KCHUNK, 256, 0, stream>>>(k, FcT, FsT, XkP);
  k_reduce<<<1024, 256, 0, stream>>>(XqP, XkP, Xq, Xk);
  k_modes<<<NBH, 256, 0, stream>>>(Xq, Xk, w1, w2, Yh);
  k_inv_dft<<<NBH * 128, 256, 0, stream>>>(Yh, GT, out);
}
